// AttLSTMCount_85246510891503
// MI455X (gfx1250) — matrix operand order repaired and checked by reading, not compiled, not run on hardware
//
#include <hip/hip_runtime.h>

typedef __attribute__((ext_vector_type(16))) __bf16 v16bf;
typedef __attribute__((ext_vector_type(8)))  __bf16 v8bf;
typedef __attribute__((ext_vector_type(8)))  float  v8f;

#define BB   128   // batch
#define TT   30    // timesteps
#define VV   10000 // vocab
#define HH   512   // hidden
#define EE   512   // embed
#define AA   512   // att dim
#define FF   2048  // feat dim
#define CDD  128   // count dim
#define CVV  64    // count vec
#define NP   49    // 7x7 positions
#define KX   3200  // E + F + CD + H
#define G4   2048  // 4*H gates
#define VPAD 10048 // vocab padded to 64

__device__ __forceinline__ unsigned short f2bf(float x) {
  unsigned u = __builtin_bit_cast(unsigned, x);
  unsigned r = ((u >> 16) & 1u) + 0x7fffu;   // round-to-nearest-even
  return (unsigned short)((u + r) >> 16);
}
__device__ __forceinline__ float sigf(float x) { return 1.f / (1.f + __expf(-x)); }

// ---------------------------------------------------------------------------
// Generic bf16 WMMA GEMM:  C[M,N] = A[M,K] * B[N,K]^T + bias[N]
// One wave (32 lanes) per block computes a 16x64 tile via 4x v_wmma_f32_16x16x32_bf16.
// A-operand layout (16-bit A 16x32, contiguous k): lane<16 holds K 0..15,
// lane>=16 holds K 16..31 (same order as B). B-operand: lane half selects K 0..15 / 16..31.
// Requires M%16==0, K%32==0, B has >=gridDim.x*64 rows (padded), 16B-aligned rows.
// ---------------------------------------------------------------------------
__global__ __launch_bounds__(32)
void wmma_gemm_bf16(const unsigned short* __restrict__ A,
                    const unsigned short* __restrict__ Bw,
                    const float* __restrict__ bias,
                    float* __restrict__ C,
                    int N, int K, int lda, int ldb, int ldc)
{
  const int lane = threadIdx.x;
  const int half = lane >> 4;
  const int r16  = lane & 15;
  const long m0  = (long)blockIdx.y * 16;
  const long n0  = (long)blockIdx.x * 64;

  const unsigned short* arow = A  + (m0 + r16) * (long)lda;
  const unsigned short* b0   = Bw + (n0 + r16) * (long)ldb;
  const unsigned short* b1   = b0 + 16L * ldb;
  const unsigned short* b2   = b0 + 32L * ldb;
  const unsigned short* b3   = b0 + 48L * ldb;
  const int ka = half * 16;   // A: element offset of contiguous 16-elt chunk
  const int kb = half * 16;   // B: element offset of contiguous 16-elt chunk

  v8f acc0 = {}, acc1 = {}, acc2 = {}, acc3 = {};
  for (int kk = 0; kk < K; kk += 32) {
    __builtin_prefetch((const void*)(arow + kk + 64 + ka), 0, 1);
    __builtin_prefetch((const void*)(b0   + kk + 64 + kb), 0, 1);
    v8bf alo = *(const v8bf*)(arow + kk + ka);
    v8bf ahi = *(const v8bf*)(arow + kk + 8 + ka);
    v16bf a = __builtin_shufflevector(alo, ahi,
              0,1,2,3,4,5,6,7,8,9,10,11,12,13,14,15);
    v16bf bm0 = *(const v16bf*)(b0 + kk + kb);
    v16bf bm1 = *(const v16bf*)(b1 + kk + kb);
    v16bf bm2 = *(const v16bf*)(b2 + kk + kb);
    v16bf bm3 = *(const v16bf*)(b3 + kk + kb);
    acc0 = __builtin_amdgcn_wmma_f32_16x16x32_bf16(false, a, false, bm0, (short)0, acc0, false, false);
    acc1 = __builtin_amdgcn_wmma_f32_16x16x32_bf16(false, a, false, bm1, (short)0, acc1, false, false);
    acc2 = __builtin_amdgcn_wmma_f32_16x16x32_bf16(false, a, false, bm2, (short)0, acc2, false, false);
    acc3 = __builtin_amdgcn_wmma_f32_16x16x32_bf16(false, a, false, bm3, (short)0, acc3, false, false);
  }

  // C/D layout: VGPR r, lane<16 -> (m0+r, n0+lane); lane>=16 -> (m0+8+r, n0+lane-16)
  const long mbase = m0 + half * 8;
  #pragma unroll
  for (int j = 0; j < 4; ++j) {
    v8f acc = (j == 0) ? acc0 : (j == 1) ? acc1 : (j == 2) ? acc2 : acc3;
    long n = n0 + 16 * j + r16;
    if (n < N) {
      float bs = bias[n];
      float* cp = C + mbase * (long)ldc + n;
      #pragma unroll
      for (int r = 0; r < 8; ++r) cp[(long)r * ldc] = acc[r] + bs;
    }
  }
}

// ---------------------------- conversion kernels ---------------------------
__global__ void k_f32_to_bf16(unsigned short* d, const float* s, long n) {
  long i = (long)blockIdx.x * 256 + threadIdx.x;
  if (i < n) d[i] = f2bf(s[i]);
}
__global__ void k_build_wfull(unsigned short* d, const float* wih, const float* whh) {
  long i = (long)blockIdx.x * 256 + threadIdx.x;
  if (i >= (long)G4 * KX) return;
  int n = (int)(i / KX), k = (int)(i % KX);
  float v = (k < EE + FF + CDD) ? wih[(long)n * (EE + FF + CDD) + k]
                                : whh[(long)n * HH + (k - (EE + FF + CDD))];
  d[i] = f2bf(v);
}
__global__ void k_fcw_pad(unsigned short* d, const float* fw) {
  long i = (long)blockIdx.x * 256 + threadIdx.x;
  if (i >= (long)VPAD * HH) return;
  int n = (int)(i / HH), k = (int)(i % HH);
  d[i] = (n < VV) ? f2bf(fw[(long)n * HH + k]) : (unsigned short)0;
}
__global__ void k_att_tr(unsigned short* d, const float* att) {
  long i = (long)blockIdx.x * 256 + threadIdx.x;
  if (i >= (long)BB * NP * FF) return;
  long b = i / (NP * FF); long r = i % (NP * FF);
  int p = (int)(r / FF), f = (int)(r % FF);
  d[i] = f2bf(att[(b * FF + f) * NP + p]);   // (B,F,49) -> (B,49,F)
}
__global__ void k_embed(unsigned short* d, const float* ew, const int* cap) {
  long i = (long)blockIdx.x * 256 + threadIdx.x;
  if (i >= (long)BB * TT * EE) return;
  long b = i / (TT * EE); long r = i % (TT * EE);
  int t = (int)(r / EE), e = (int)(r % EE);
  int tok = cap[b * 31 + t];
  d[i] = f2bf(ew[(long)tok * EE + e]);
}
__global__ void k_bias_full(float* d, const float* bih, const float* bhh) {
  int i = blockIdx.x * 256 + threadIdx.x;
  if (i < G4) d[i] = bih[i] + bhh[i];
}

// count_embed = relu(cv @ w1^T + b1) @ w2^T + b2   (tiny, one block per b)
__global__ __launch_bounds__(CDD)
void k_count(const float* cv, const float* w1, const float* b1,
             const float* w2, const float* b2, float* out) {
  int b = blockIdx.x, n = threadIdx.x;
  __shared__ float cvs[CVV];
  __shared__ float h1[CDD];
  if (n < CVV) cvs[n] = cv[b * CVV + n];
  __syncthreads();
  float s = b1[n];
  for (int k = 0; k < CVV; ++k) s += cvs[k] * w1[n * CVV + k];
  h1[n] = fmaxf(s, 0.f);
  __syncthreads();
  float o = b2[n];
  for (int k = 0; k < CDD; ++k) o += h1[k] * w2[n * CDD + k];
  out[b * CDD + n] = o;
}

__global__ void k_h0(const float* tmp, unsigned short* h_bf, float* c) {
  int i = blockIdx.x * 256 + threadIdx.x;   // B*H
  h_bf[i] = f2bf(tanhf(tmp[i]));
  c[i] = 0.f;
}

__global__ void k_lstm(const float* gates, float* c, unsigned short* h_bf) {
  int i = blockIdx.x * 256 + threadIdx.x;   // B*H
  int b = i >> 9, n = i & 511;
  const float* g = gates + (long)b * G4;
  float ig = sigf(g[n]);
  float fg = sigf(g[n + 512]);
  float gg = tanhf(g[n + 1024]);
  float og = sigf(g[n + 1536]);
  float cn = fg * c[i] + ig * gg;
  c[i] = cn;
  h_bf[i] = f2bf(og * tanhf(cn));
}

// attention: scores/softmax/context, then assemble x = [we_t|context|cemb|h]
__global__ __launch_bounds__(256)
void k_attention(const float* __restrict__ hp, const float* __restrict__ att_proj,
                 const float* __restrict__ alpha_w, const float* __restrict__ alpha_b,
                 const float* __restrict__ att_feats, const float* __restrict__ cemb,
                 const unsigned short* __restrict__ we, const unsigned short* __restrict__ h_bf,
                 unsigned short* __restrict__ x, float* __restrict__ alphas, int t)
{
  const int b = blockIdx.x, tid = threadIdx.x;
  __shared__ float hp_s[AA], aw_s[AA], sc_s[NP], al_s[NP], inv_s;
  for (int i = tid; i < AA; i += 256) { hp_s[i] = hp[b * AA + i]; aw_s[i] = alpha_w[i]; }
  __syncthreads();
  const int wv = tid >> 5, ln = tid & 31;
  for (int j = wv; j < NP; j += 8) {
    const float* ap = att_proj + ((long)b * NP + j) * AA;
    float p = 0.f;
    for (int a2 = ln; a2 < AA; a2 += 32) p += tanhf(ap[a2] + hp_s[a2]) * aw_s[a2];
    for (int o = 16; o > 0; o >>= 1) p += __shfl_down(p, o, 32);
    if (ln == 0) sc_s[j] = p + alpha_b[0];
  }
  __syncthreads();
  if (tid == 0) {
    float mx = sc_s[0];
    for (int j = 1; j < NP; ++j) mx = fmaxf(mx, sc_s[j]);
    float s = 0.f;
    for (int j = 0; j < NP; ++j) { float e = __expf(sc_s[j] - mx); al_s[j] = e; s += e; }
    inv_s = 1.f / s;
  }
  __syncthreads();
  const float inv = inv_s;
  unsigned short* xb = x + (long)b * KX;
  if (tid < NP) alphas[((long)b * TT + t) * NP + tid] = al_s[tid] * inv;
  for (int i = tid; i < EE; i += 256) xb[i] = we[((long)b * TT + t) * EE + i];
  const float* af = att_feats + (long)b * FF * NP;
  for (int f = tid; f < FF; f += 256) {
    const float* rowp = af + (long)f * NP;
    float acc = 0.f;
    for (int j = 0; j < NP; ++j) acc += al_s[j] * rowp[j];
    xb[EE + f] = f2bf(acc * inv);
  }
  if (tid < CDD) xb[EE + FF + tid] = f2bf(cemb[b * CDD + tid]);
  for (int i = tid; i < HH; i += 256) xb[EE + FF + CDD + i] = h_bf[b * HH + i];
}

// ---------------------------------------------------------------------------
extern "C" void kernel_launch(void* const* d_in, const int* in_sizes, int n_in,
                              void* d_out, int out_size, void* d_ws, size_t ws_size,
                              hipStream_t stream)
{
  const float* att_feats  = (const float*)d_in[0];
  const float* fc_feats   = (const float*)d_in[1];
  const int*   captions   = (const int*)  d_in[2];
  const float* count_vecs = (const float*)d_in[3];
  const float* embed_w    = (const float*)d_in[4];
  const float* cm_w1      = (const float*)d_in[5];
  const float* cm_b1      = (const float*)d_in[6];
  const float* cm_w2      = (const float*)d_in[7];
  const float* cm_b2      = (const float*)d_in[8];
  const float* att_hw     = (const float*)d_in[9];
  const float* att_hb     = (const float*)d_in[10];
  const float* att_fw     = (const float*)d_in[11];
  const float* att_fb     = (const float*)d_in[12];
  const float* alpha_w    = (const float*)d_in[13];
  const float* alpha_b    = (const float*)d_in[14];
  const float* fcproj_w   = (const float*)d_in[15];
  const float* fcproj_b   = (const float*)d_in[16];
  const float* w_ih       = (const float*)d_in[17];
  const float* b_ih       = (const float*)d_in[18];
  const float* w_hh       = (const float*)d_in[19];
  const float* b_hh       = (const float*)d_in[20];
  const float* fc_w       = (const float*)d_in[21];
  const float* fc_b       = (const float*)d_in[22];
  (void)in_sizes; (void)n_in; (void)out_size; (void)ws_size;

  char* wp = (char*)d_ws;
  auto alloc = [&](size_t bytes) -> void* {
    void* p = (void*)wp; wp += (bytes + 255) & ~(size_t)255; return p;
  };
  unsigned short* Wfull   = (unsigned short*)alloc((size_t)G4 * KX * 2);    // [w_ih|w_hh] bf16
  unsigned short* fcwPad  = (unsigned short*)alloc((size_t)VPAD * HH * 2);  // fc_w padded
  unsigned short* attHw   = (unsigned short*)alloc((size_t)AA * HH * 2);
  unsigned short* fcprojW = (unsigned short*)alloc((size_t)HH * FF * 2);
  unsigned short* attFw   = (unsigned short*)alloc((size_t)AA * FF * 2);
  unsigned short* fcFeats = (unsigned short*)alloc((size_t)BB * FF * 2);
  unsigned short* attTr   = (unsigned short*)alloc((size_t)BB * NP * FF * 2);
  unsigned short* weBf    = (unsigned short*)alloc((size_t)BB * TT * EE * 2);
  float* attProj  = (float*)alloc((size_t)BB * NP * AA * 4);
  float* cEmb     = (float*)alloc((size_t)BB * CDD * 4);
  float* biasFull = (float*)alloc((size_t)G4 * 4);
  float* hpBuf    = (float*)alloc((size_t)BB * AA * 4);
  float* gates    = (float*)alloc((size_t)BB * G4 * 4);   // also reused as h0 tmp
  unsigned short* xFull = (unsigned short*)alloc((size_t)BB * KX * 2);
  unsigned short* hBf   = (unsigned short*)alloc((size_t)BB * HH * 2);
  float* cState   = (float*)alloc((size_t)BB * HH * 4);

  float* outLogits = (float*)d_out;                        // (B, T, V)
  float* outAlphas = outLogits + (size_t)BB * TT * VV;     // (B, T, 49)

  auto nb = [](long n) { return (unsigned)((n + 255) / 256); };

  // ---- one-time setup: bf16 weight/activation staging ----
  k_build_wfull<<<nb((long)G4 * KX), 256, 0, stream>>>(Wfull, w_ih, w_hh);
  k_fcw_pad    <<<nb((long)VPAD * HH), 256, 0, stream>>>(fcwPad, fc_w);
  k_f32_to_bf16<<<nb((long)AA * HH), 256, 0, stream>>>(attHw, att_hw, (long)AA * HH);
  k_f32_to_bf16<<<nb((long)HH * FF), 256, 0, stream>>>(fcprojW, fcproj_w, (long)HH * FF);
  k_f32_to_bf16<<<nb((long)AA * FF), 256, 0, stream>>>(attFw, att_fw, (long)AA * FF);
  k_f32_to_bf16<<<nb((long)BB * FF), 256, 0, stream>>>(fcFeats, fc_feats, (long)BB * FF);
  k_att_tr     <<<nb((long)BB * NP * FF), 256, 0, stream>>>(attTr, att_feats);
  k_embed      <<<nb((long)BB * TT * EE), 256, 0, stream>>>(weBf, embed_w, captions);
  k_bias_full  <<<nb(G4), 256, 0, stream>>>(biasFull, b_ih, b_hh);
  k_count      <<<BB, CDD, 0, stream>>>(count_vecs, cm_w1, cm_b1, cm_w2, cm_b2, cEmb);

  // h0 = tanh(fc_feats @ fcproj_w^T + b) ; c0 = 0     (128x512x2048 WMMA)
  { dim3 g(HH / 64, BB / 16);
    wmma_gemm_bf16<<<g, 32, 0, stream>>>(fcFeats, fcprojW, fcproj_b, gates, HH, FF, FF, FF, HH); }
  k_h0<<<nb((long)BB * HH), 256, 0, stream>>>(gates, hBf, cState);

  // att_proj = att @ att_fw^T + b                      (6272x512x2048 WMMA)
  { dim3 g(AA / 64, (BB * NP) / 16);
    wmma_gemm_bf16<<<g, 32, 0, stream>>>(attTr, attFw, att_fb, attProj, AA, FF, FF, FF, AA); }

  // ---- sequential timesteps ----
  for (int t = 0; t < TT; ++t) {
    // hp = h @ att_hw^T + b                            (128x512x512 WMMA)
    { dim3 g(AA / 64, BB / 16);
      wmma_gemm_bf16<<<g, 32, 0, stream>>>(hBf, attHw, att_hb, hpBuf, AA, HH, HH, HH, AA); }
    // attention + softmax + context + assemble x
    k_attention<<<BB, 256, 0, stream>>>(hpBuf, attProj, alpha_w, alpha_b,
                                        att_feats, cEmb, weBf, hBf, xFull, outAlphas, t);
    // gates = x @ [w_ih|w_hh]^T + (b_ih + b_hh)        (128x2048x3200 WMMA)
    { dim3 g(G4 / 64, BB / 16);
      wmma_gemm_bf16<<<g, 32, 0, stream>>>(xFull, Wfull, biasFull, gates, G4, KX, KX, KX, G4); }
    // LSTM cell pointwise
    k_lstm<<<nb((long)BB * HH), 256, 0, stream>>>(gates, cState, hBf);
    // logits = h_new @ fc_w^T + fc_b                   (128x10000x512 WMMA)
    { dim3 g(VPAD / 64, BB / 16);
      wmma_gemm_bf16<<<g, 32, 0, stream>>>(hBf, fcwPad, fc_b,
                                           outLogits + (size_t)t * VV, VV, HH, HH, HH, TT * VV); }
  }
}
